// ResidentialTransMat_21680994910755
// MI455X (gfx1250) — compile-verified, matrix operand order checked by reading, not run on hardware
//
#include <hip/hip_runtime.h>
#include <hip/hip_bf16.h>

// Problem sizes (fixed by the reference)
#define NSTATE 2048   // S: softmax rows, GEMM K and N
#define NBATCH 4096   // B: GEMM M
#define SM1    2047

typedef unsigned short u16;
typedef __attribute__((ext_vector_type(16))) __bf16 v16bf;
typedef __attribute__((ext_vector_type(8)))  float  v8f;

union FragBF {
  v16bf v;
  u16   h[16];
  uint4 q[2];
};

// Exact fp32 -> bf16_hi + bf16_lo split (truncation split: hi exactly
// representable, f - hi exact, lo truncated -> ~2^-17 total rel. error).
__device__ __forceinline__ void split_bf16(float f, u16& hi, u16& lo) {
  unsigned u  = __float_as_uint(f);
  unsigned hb = u & 0xFFFF0000u;
  hi = (u16)(hb >> 16);
  float l = f - __uint_as_float(hb);
  lo = (u16)(__float_as_uint(l) >> 16);
}

// ---------------------------------------------------------------------------
// Phase 1a: per-row softmax stats of U (2048 x 2047): rowMax[s], rowInv[s].
// ---------------------------------------------------------------------------
__global__ __launch_bounds__(256)
void rowstats_kernel(const float* __restrict__ U,
                     float* __restrict__ rowMax, float* __restrict__ rowInv) {
  const int s   = blockIdx.x;
  const int tid = threadIdx.x;
  const float* row = U + (size_t)s * SM1;

  __shared__ float red[256];

  float m = -3.4e38f;
  for (int j = tid; j < SM1; j += 256) m = fmaxf(m, row[j]);
  red[tid] = m;
  __syncthreads();
  for (int off = 128; off > 0; off >>= 1) {
    if (tid < off) red[tid] = fmaxf(red[tid], red[tid + off]);
    __syncthreads();
  }
  m = red[0];
  __syncthreads();

  float sum = 0.0f;
  for (int j = tid; j < SM1; j += 256) sum += __expf(row[j] - m);
  red[tid] = sum;
  __syncthreads();
  for (int off = 128; off > 0; off >>= 1) {
    if (tid < off) red[tid] += red[tid + off];
    __syncthreads();
  }
  if (tid == 0) {
    rowMax[s] = m;
    rowInv[s] = 1.0f / red[0];
  }
}

// ---------------------------------------------------------------------------
// Phase 1b: build the TRANSPOSED, bf16-hi/lo-split transition matrix:
//   matT_hi[t][s] + matT_lo[t][s] ~= mat[s][t]
//   mat[s][t] = (t==s) ? 0 : exp(U[s][t-(t>s)] - rowMax[s]) * rowInv[s]
// 64x64 tile transpose through LDS; coalesced reads (along t) and
// coalesced writes (along s).
// ---------------------------------------------------------------------------
__global__ __launch_bounds__(256)
void build_matT_kernel(const float* __restrict__ U,
                       const float* __restrict__ rowMax,
                       const float* __restrict__ rowInv,
                       u16* __restrict__ TH, u16* __restrict__ TL) {
  __shared__ unsigned tile[64 * 68];   // packed (hi<<16)|lo, padded rows

  const int tid = threadIdx.x;
  const int t0  = blockIdx.x * 64;     // t range of this tile
  const int s0  = blockIdx.y * 64;     // s range of this tile

  // pass 1: compute mat[s][t] for the tile, store packed into LDS[s][t]
#pragma unroll
  for (int p = 0; p < 4; ++p) {
    const int sl = (tid >> 4) + p * 16;      // 0..63
    const int gs = s0 + sl;
    const float rm = rowMax[gs];
    const float ri = rowInv[gs];
    const int tl0 = (tid & 15) * 4;          // 4 consecutive t
#pragma unroll
    for (int q = 0; q < 4; ++q) {
      const int gt = t0 + tl0 + q;
      unsigned packed = 0;
      if (gt != gs) {
        const int j = gt - (gt > gs ? 1 : 0);
        const float pr = __expf(U[(size_t)gs * SM1 + j] - rm) * ri;
        u16 hi, lo;
        split_bf16(pr, hi, lo);
        packed = ((unsigned)hi << 16) | (unsigned)lo;
      }
      tile[sl * 68 + tl0 + q] = packed;
    }
  }
  __syncthreads();

  // pass 2: write transposed rows (t-major, contiguous in s)
#pragma unroll
  for (int p = 0; p < 4; ++p) {
    const int tl = (tid >> 4) + p * 16;      // 0..63 (output row)
    const int sl0 = (tid & 15) * 4;          // 4 consecutive s
    u16 h4[4], l4[4];
#pragma unroll
    for (int q = 0; q < 4; ++q) {
      const unsigned packed = tile[(sl0 + q) * 68 + tl];
      h4[q] = (u16)(packed >> 16);
      l4[q] = (u16)(packed & 0xFFFFu);
    }
    const size_t o = (size_t)(t0 + tl) * NSTATE + s0 + sl0;
    *(uint2*)(TH + o) = *(const uint2*)h4;
    *(uint2*)(TL + o) = *(const uint2*)l4;
  }
}

// ---------------------------------------------------------------------------
// Phase 1c: split alpha (4096 x 2048 fp32) into bf16 hi/lo arrays.
// ---------------------------------------------------------------------------
__global__ __launch_bounds__(256)
void split_alpha_kernel(const float* __restrict__ A,
                        u16* __restrict__ AH, u16* __restrict__ AL) {
  const size_t base = ((size_t)blockIdx.x * 256 + threadIdx.x) * 8;
  u16 h8[8], l8[8];
#pragma unroll
  for (int i = 0; i < 8; ++i) split_bf16(A[base + i], h8[i], l8[i]);
  *(uint4*)(AH + base) = *(const uint4*)h8;
  *(uint4*)(AL + base) = *(const uint4*)l8;
}

// ---------------------------------------------------------------------------
// Phase 2: out(4096 x 2048) = alpha @ mat via bf16x3 WMMA, operands pre-split.
//
// Block tile: 128(M) x 256(N), 256 threads = 8 waves as 2(M) x 4(N).
// Wave tile:  64 x 64 = 4 x 4 fragments (v8f accumulators, 128 VGPRs).
// K step: 32. A staged as [m][k] bf16, B staged as [n][k] bf16 (from matT),
// so every fragment is exactly two aligned ds_load_b128 — no repacking VALU.
// ---------------------------------------------------------------------------
#define LDK 40   // bf16 elements per LDS row: 32 + 8 pad, 80B rows (16B-aligned)

__global__ __launch_bounds__(256)
void gemm_bf16x3_wmma_kernel(const u16* __restrict__ AH, const u16* __restrict__ AL,
                             const u16* __restrict__ BH, const u16* __restrict__ BL,
                             float* __restrict__ C) {
  __shared__ u16 sA[2][128 * LDK];   // [hi/lo][m][k]  (20 KB)
  __shared__ u16 sB[2][256 * LDK];   // [hi/lo][n][k]  (40 KB)

  const int tid  = threadIdx.x;
  const int lane = tid & 31;
  const int wave = tid >> 5;
  const int wm   = wave >> 2;    // 0..1
  const int wn   = wave & 3;     // 0..3
  const int bm   = blockIdx.y;   // 0..31
  const int bn   = blockIdx.x;   // 0..7

  const int l16  = lane & 15;
  const int half = lane >> 4;

  v8f acc[4][4];
#pragma unroll
  for (int i = 0; i < 4; ++i)
#pragma unroll
    for (int j = 0; j < 4; ++j) {
      v8f z = {};
      acc[i][j] = z;
    }

  for (int k0 = 0; k0 < NSTATE; k0 += 32) {
    // --- stage A tiles (hi & lo): 128 rows x 32 bf16 = 512 b128 slots each
#pragma unroll
    for (int i = 0; i < 2; ++i) {
      const int slot = tid + i * 256;          // 0..511
      const int r    = slot >> 2;              // 0..127
      const int c    = (slot & 3) << 3;        // 0,8,16,24
      const size_t g = (size_t)(bm * 128 + r) * NSTATE + k0 + c;
      *(uint4*)(&sA[0][r * LDK + c]) = *(const uint4*)(AH + g);
      *(uint4*)(&sA[1][r * LDK + c]) = *(const uint4*)(AL + g);
    }
    // --- stage B tiles (hi & lo): 256 rows x 32 bf16 = 1024 b128 slots each
#pragma unroll
    for (int i = 0; i < 4; ++i) {
      const int slot = tid + i * 256;          // 0..1023
      const int r    = slot >> 2;              // 0..255
      const int c    = (slot & 3) << 3;        // 0,8,16,24
      const size_t g = (size_t)(bn * 256 + r) * NSTATE + k0 + c;
      *(uint4*)(&sB[0][r * LDK + c]) = *(const uint4*)(BH + g);
      *(uint4*)(&sB[1][r * LDK + c]) = *(const uint4*)(BL + g);
    }
    __syncthreads();

    // --- B fragments for this wave's four 16-wide N slices (lane = N, elem = K)
    FragBF bhi[4], blo[4];
#pragma unroll
    for (int j = 0; j < 4; ++j) {
      const u16* rb = &sB[0][(wn * 64 + j * 16 + l16) * LDK];
      const u16* rl = &sB[1][(wn * 64 + j * 16 + l16) * LDK];
      bhi[j].q[0] = *(const uint4*)(rb + half * 8);
      bhi[j].q[1] = *(const uint4*)(rb + 16 + half * 8);
      blo[j].q[0] = *(const uint4*)(rl + half * 8);
      blo[j].q[1] = *(const uint4*)(rl + 16 + half * 8);
    }

    // --- A fragments + 12 WMMAs per M slice
#pragma unroll
    for (int i = 0; i < 4; ++i) {
      const u16* ra = &sA[0][(wm * 64 + i * 16 + l16) * LDK];
      const u16* rl = &sA[1][(wm * 64 + i * 16 + l16) * LDK];
      FragBF ahi, alo;
      ahi.q[0] = *(const uint4*)(ra + half * 8);
      ahi.q[1] = *(const uint4*)(ra + 16 + half * 8);
      alo.q[0] = *(const uint4*)(rl + half * 8);
      alo.q[1] = *(const uint4*)(rl + 16 + half * 8);

#pragma unroll
      for (int j = 0; j < 4; ++j) {
        acc[i][j] = __builtin_amdgcn_wmma_f32_16x16x32_bf16(
            false, ahi.v, false, bhi[j].v, (short)0, acc[i][j], false, false);
        acc[i][j] = __builtin_amdgcn_wmma_f32_16x16x32_bf16(
            false, ahi.v, false, blo[j].v, (short)0, acc[i][j], false, false);
        acc[i][j] = __builtin_amdgcn_wmma_f32_16x16x32_bf16(
            false, alo.v, false, bhi[j].v, (short)0, acc[i][j], false, false);
      }
    }
    __syncthreads();
  }

  // --- epilogue: C fragment layout: VGPR r -> m = r + 8*half, n = l16
#pragma unroll
  for (int i = 0; i < 4; ++i) {
    const int row0 = bm * 128 + wm * 64 + i * 16 + half * 8;
#pragma unroll
    for (int j = 0; j < 4; ++j) {
      const int col0 = bn * 256 + wn * 64 + j * 16 + l16;
      float* out = C + (size_t)row0 * NSTATE + col0;
#pragma unroll
      for (int r = 0; r < 8; ++r)
        out[(size_t)r * NSTATE] = acc[i][j][r];
    }
  }
}

extern "C" void kernel_launch(void* const* d_in, const int* in_sizes, int n_in,
                              void* d_out, int out_size, void* d_ws, size_t ws_size,
                              hipStream_t stream) {
  (void)in_sizes; (void)n_in; (void)out_size; (void)ws_size;
  // setup_inputs order: 0=state_embeddings (unused), 1=alpha, 2=context (unused),
  //                     3=unnormalized_transition_matrix
  const float* alpha  = (const float*)d_in[1];
  const float* unnorm = (const float*)d_in[3];
  float* out = (float*)d_out;

  // workspace layout (~50.4 MB total)
  u16* aHi = (u16*)d_ws;                         // 16 MB
  u16* aLo = aHi + (size_t)NBATCH * NSTATE;      // 16 MB
  u16* tHi = aLo + (size_t)NBATCH * NSTATE;      //  8 MB
  u16* tLo = tHi + (size_t)NSTATE * NSTATE;      //  8 MB
  float* rMax = (float*)(tLo + (size_t)NSTATE * NSTATE);
  float* rInv = rMax + NSTATE;

  rowstats_kernel<<<NSTATE, 256, 0, stream>>>(unnorm, rMax, rInv);
  build_matT_kernel<<<dim3(NSTATE / 64, NSTATE / 64), 256, 0, stream>>>(
      unnorm, rMax, rInv, tHi, tLo);
  split_alpha_kernel<<<(NBATCH * NSTATE) / (256 * 8), 256, 0, stream>>>(
      alpha, aHi, aLo);

  dim3 grid(NSTATE / 256, NBATCH / 128);   // (8, 32)
  gemm_bf16x3_wmma_kernel<<<grid, 256, 0, stream>>>(aHi, aLo, tHi, tLo, out);
}
